// PyTorchSSM_55319178773119
// MI455X (gfx1250) — compile-verified, matrix operand order checked by reading, not run on hardware
//
#include <hip/hip_runtime.h>
#include <hip/hip_bf16.h>

// ---------------- model constants ----------------
enum : int {
  D_MODEL  = 1024,
  D_STATE  = 128,
  D_CONV   = 4,
  HEADDIM  = 64,
  D_INNER  = 2048,               // EXPAND * D_MODEL
  NHEADS   = 32,                 // D_INNER / HEADDIM
  CONV_DIM = D_INNER + 2 * D_STATE,          // 2304
  D_IN_PROJ = 2 * D_INNER + 2 * D_STATE + NHEADS, // 4384
  BATCH = 2,
  SEQ   = 4096,
  ROWS  = BATCH * SEQ            // 8192
};

typedef __attribute__((ext_vector_type(16))) __bf16 v16bf;
typedef __attribute__((ext_vector_type(8)))  float  v8f;
typedef int async_v4i __attribute__((vector_size(16)));   // matches builtin param type

union FragCast {
  uint4 q[2];
  v16bf v;
};

__device__ __forceinline__ unsigned short f2bf(float f) {
  unsigned int u = __float_as_uint(f);
  unsigned int r = u + 0x7FFFu + ((u >> 16) & 1u);   // RNE
  return (unsigned short)(r >> 16);
}

__device__ __forceinline__ float sigm(float x) {
  return 1.0f / (1.0f + __expf(-x));
}

// Async global->LDS copy availability (CDNA5 ASYNCcnt path)
#if defined(__gfx1250__) && __has_builtin(__builtin_amdgcn_global_load_async_to_lds_b128)
#define USE_ASYNC_LDS 1
#else
#define USE_ASYNC_LDS 0
#endif

__device__ __forceinline__ void wait_async_all() {
#if USE_ASYNC_LDS
#if __has_builtin(__builtin_amdgcn_s_wait_asynccnt)
  __builtin_amdgcn_s_wait_asynccnt(0);
#else
  asm volatile("s_wait_asynccnt 0" ::: "memory");
#endif
#endif
}

// ---------------- fp32 -> bf16 conversion ----------------
__global__ void cvt_bf16_kernel(const float* __restrict__ in,
                                unsigned short* __restrict__ out, int n) {
  int i = blockIdx.x * 256 + threadIdx.x;
  if (i < n) out[i] = f2bf(in[i]);
}

// ---------------- bf16 WMMA GEMM: C(MxN,f32) = A(MxK) * B(KxN) ----------------
// Row-major A and B (bf16 as ushort). Block tile 128x128, BK=32.
// 8 waves; wave (wm in 0..1, wn in 0..3) owns 64x32 => 4x2 tiles of 16x16.
// Requires: M % 128 == 0, K % 32 == 0, N % 16 == 0.
__global__ __launch_bounds__(256)
void gemm_bf16_wmma(const unsigned short* __restrict__ A,
                    const unsigned short* __restrict__ Bw,
                    float* __restrict__ C,
                    int M, int N, int K) {
  __shared__ __align__(16) unsigned short lsA[128 * 32];  // [row][k]
  __shared__ __align__(16) unsigned short lsB[128 * 32];  // transposed: [n][k]

  const int tid  = threadIdx.x;
  const int lane = tid & 31;
  const int wave = tid >> 5;
  const int wm   = wave & 1;   // 0..1  (row block of 64)
  const int wn   = wave >> 1;  // 0..3  (col block of 32)
  const int blockM = blockIdx.y * 128;
  const int blockN = blockIdx.x * 128;

  const int rowInLane = lane & 15;
  const int halfK     = lane >> 4;   // 0/1 -> K offset +8

  v8f acc[4][2];
#pragma unroll
  for (int i = 0; i < 4; ++i)
#pragma unroll
    for (int j = 0; j < 2; ++j)
#pragma unroll
      for (int r = 0; r < 8; ++r) acc[i][j][r] = 0.0f;

  // staging decomposition
  const int aRow     = tid >> 1;        // 0..127
  const int aColHalf = tid & 1;         // 16 elements each
  const int bK  = tid >> 3;             // 0..31
  const int bN0 = (tid & 7) * 16;       // 0..112

  for (int kb = 0; kb < K; kb += 32) {
    // ---- stage A tile (128x32): async global->LDS (bypasses VGPRs) ----
    {
      const unsigned short* gsrc = A + (size_t)(blockM + aRow) * K + kb + aColHalf * 16;
      unsigned short* ldst = &lsA[aRow * 32 + aColHalf * 16];
#if USE_ASYNC_LDS
      __builtin_amdgcn_global_load_async_to_lds_b128(
          (__attribute__((address_space(1))) async_v4i*)gsrc,
          (__attribute__((address_space(3))) async_v4i*)ldst, 0, 0);
      __builtin_amdgcn_global_load_async_to_lds_b128(
          (__attribute__((address_space(1))) async_v4i*)(gsrc + 8),
          (__attribute__((address_space(3))) async_v4i*)(ldst + 8), 0, 0);
#else
      const uint4* src = (const uint4*)gsrc;
      uint4 v0 = src[0];
      uint4 v1 = src[1];
      uint4* dst = (uint4*)ldst;
      dst[0] = v0;
      dst[1] = v1;
#endif
    }
    // ---- stage B tile (32x128), transposed into LDS as [n][k] ----
    {
      unsigned short tmp[16];
      int gN = blockN + bN0;
      if (gN < N) {
        const uint4* src = (const uint4*)(Bw + (size_t)(kb + bK) * N + gN);
        uint4 v0 = src[0];
        uint4 v1 = src[1];
        *(uint4*)&tmp[0] = v0;
        *(uint4*)&tmp[8] = v1;
      } else {
#pragma unroll
        for (int i = 0; i < 16; ++i) tmp[i] = 0;
      }
#pragma unroll
      for (int i = 0; i < 16; ++i) lsB[(bN0 + i) * 32 + bK] = tmp[i];
    }
    wait_async_all();   // drain ASYNCcnt before the workgroup barrier
    __syncthreads();

    // prefetch next K tiles into L2 (global_prefetch_b8)
    if (kb + 32 < K) {
      __builtin_prefetch(A + (size_t)(blockM + aRow) * K + (kb + 32) + aColHalf * 16, 0, 3);
      int gN = blockN + bN0;
      if (gN < N) __builtin_prefetch(Bw + (size_t)(kb + 32 + bK) * N + gN, 0, 3);
    }

    // ---- load fragments per documented 16-bit A layout ----
    FragCast afr[4], bfr[2];
#pragma unroll
    for (int mt = 0; mt < 4; ++mt) {
      int r = wm * 64 + mt * 16 + rowInLane;
      const unsigned short* base = &lsA[r * 32 + halfK * 8];
      afr[mt].q[0] = *(const uint4*)(base);       // K = halfK*8 .. +7
      afr[mt].q[1] = *(const uint4*)(base + 16);  // K = 16 + halfK*8 .. +7
    }
#pragma unroll
    for (int nt = 0; nt < 2; ++nt) {
      int c = wn * 32 + nt * 16 + rowInLane;
      const unsigned short* base = &lsB[c * 32 + halfK * 8];
      bfr[nt].q[0] = *(const uint4*)(base);
      bfr[nt].q[1] = *(const uint4*)(base + 16);
    }

#pragma unroll
    for (int mt = 0; mt < 4; ++mt)
#pragma unroll
      for (int nt = 0; nt < 2; ++nt)
        acc[mt][nt] = __builtin_amdgcn_wmma_f32_16x16x32_bf16(
            false, afr[mt].v, false, bfr[nt].v,
            (short)0, acc[mt][nt], false, false);
    __syncthreads();
  }

  // ---- store: D layout: VGPR r -> row (halfK*8 + r), col = lane%16 ----
#pragma unroll
  for (int mt = 0; mt < 4; ++mt) {
#pragma unroll
    for (int nt = 0; nt < 2; ++nt) {
      int gCol = blockN + wn * 32 + nt * 16 + rowInLane;
      if (gCol < N) {
#pragma unroll
        for (int r = 0; r < 8; ++r) {
          int gRow = blockM + wm * 64 + mt * 16 + halfK * 8 + r;
          C[(size_t)gRow * N + gCol] = acc[mt][nt][r];
        }
      }
    }
  }
}

// ---------------- causal depthwise conv (K=4) + SiLU ----------------
// xBC lives inside zxbcdt at column offset D_INNER; output to xBC_act.
__global__ void conv_silu_kernel(const float* __restrict__ zxbcdt,
                                 const float* __restrict__ conv_w,
                                 const float* __restrict__ conv_b,
                                 float* __restrict__ xBC_act) {
  int i = blockIdx.x * 256 + threadIdx.x;
  if (i >= ROWS * CONV_DIM) return;
  int c  = i % CONV_DIM;
  int bl = i / CONV_DIM;
  int l  = bl % SEQ;

  float y = conv_b[c];
#pragma unroll
  for (int d = 0; d < D_CONV; ++d) {        // y[t] = sum_d w[d]*x[t-d]
    if (l >= d) {
      float xv = zxbcdt[(size_t)(bl - d) * D_IN_PROJ + D_INNER + c];
      y += conv_w[d * CONV_DIM + c] * xv;
    }
  }
  xBC_act[(size_t)bl * CONV_DIM + c] = y * sigm(y);
}

// ---------------- dt = softplus(dt_raw + dt_bias) ----------------
__global__ void dt_softplus_kernel(const float* __restrict__ zxbcdt,
                                   const float* __restrict__ dt_bias,
                                   float* __restrict__ dt_proc) {
  int i = blockIdx.x * 256 + threadIdx.x;
  if (i >= ROWS * NHEADS) return;
  int h   = i & (NHEADS - 1);
  int row = i >> 5;
  float v = zxbcdt[(size_t)row * D_IN_PROJ + (D_IN_PROJ - NHEADS) + h] + dt_bias[h];
  dt_proc[i] = (v > 20.0f) ? v : log1pf(__expf(v));
}

// ---------------- sequential selective-state scan ----------------
// One workgroup per (b,h). Thread t: p = t>>2 (0..63), n-slice = (t&3)*32..+32.
// State held in registers (32 f32/thread); B/C staged in LDS per step.
// Fuses the D-skip: y += D[h] * x.
__global__ __launch_bounds__(256)
void scan_kernel(const float* __restrict__ xBC_act,
                 const float* __restrict__ dt_proc,
                 const float* __restrict__ A_log,
                 const float* __restrict__ Dp,
                 float* __restrict__ y) {
  const int wg = blockIdx.x;
  const int b  = wg / NHEADS;
  const int h  = wg % NHEADS;
  const int t  = threadIdx.x;
  const int p  = t >> 2;
  const int n0 = (t & 3) * 32;

  const float A_h = -__expf(A_log[h]);
  const float D_h = Dp[h];

  float state[32];
#pragma unroll
  for (int i = 0; i < 32; ++i) state[i] = 0.0f;

  __shared__ float sB[D_STATE];
  __shared__ float sC[D_STATE];

  for (int l = 0; l < SEQ; ++l) {
    const size_t bl = (size_t)b * SEQ + l;
    if (t < D_STATE) {
      sB[t] = xBC_act[bl * CONV_DIM + D_INNER + t];
    } else if (t < 2 * D_STATE) {
      sC[t - D_STATE] = xBC_act[bl * CONV_DIM + D_INNER + D_STATE + (t - D_STATE)];
    }
    __syncthreads();

    float dt_t  = dt_proc[bl * NHEADS + h];
    float x_t   = xBC_act[bl * CONV_DIM + h * HEADDIM + p];
    float decay = __expf(A_h * dt_t);
    float xdt   = x_t * dt_t;

    float partial = 0.0f;
#pragma unroll
    for (int i = 0; i < 32; ++i) {
      int n = n0 + i;
      float s = state[i] * decay + xdt * sB[n];
      state[i] = s;
      partial += s * sC[n];
    }
    // reduce the 4 n-slices (consecutive lanes in one wave32)
    partial += __shfl_xor(partial, 1, 32);
    partial += __shfl_xor(partial, 2, 32);
    if ((t & 3) == 0) {
      y[bl * D_INNER + h * HEADDIM + p] = partial + D_h * x_t;
    }
    __syncthreads();
  }
}

// ---------------- gated RMSNorm: g = y*silu(z); g *= rsqrt(mean g^2)*norm_w; -> bf16 ----------------
__global__ __launch_bounds__(256)
void gate_rmsnorm_kernel(const float* __restrict__ y,
                         const float* __restrict__ zxbcdt,
                         const float* __restrict__ norm_w,
                         unsigned short* __restrict__ g_bf16) {
  const int row  = blockIdx.x;
  const int tid  = threadIdx.x;
  const int lane = tid & 31;
  const int wave = tid >> 5;

  float vals[8];
  float ss = 0.0f;
#pragma unroll
  for (int j = 0; j < 8; ++j) {
    int i = tid + j * 256;
    float yv = y[(size_t)row * D_INNER + i];
    float zv = zxbcdt[(size_t)row * D_IN_PROJ + i];   // z = first D_INNER cols
    float gv = yv * (zv * sigm(zv));
    vals[j] = gv;
    ss += gv * gv;
  }
#pragma unroll
  for (int off = 16; off > 0; off >>= 1) ss += __shfl_xor(ss, off, 32);

  __shared__ float red[8];
  if (lane == 0) red[wave] = ss;
  __syncthreads();
  float tot = 0.0f;
#pragma unroll
  for (int w = 0; w < 8; ++w) tot += red[w];
  float rstd = rsqrtf(tot / (float)D_INNER + 1e-5f);

#pragma unroll
  for (int j = 0; j < 8; ++j) {
    int i = tid + j * 256;
    g_bf16[(size_t)row * D_INNER + i] = f2bf(vals[j] * rstd * norm_w[i]);
  }
}

// ---------------- host launcher ----------------
static inline size_t alignUp256(size_t v) { return (v + 255) & ~(size_t)255; }

extern "C" void kernel_launch(void* const* d_in, const int* in_sizes, int n_in,
                              void* d_out, int out_size, void* d_ws, size_t ws_size,
                              hipStream_t stream) {
  (void)in_sizes; (void)n_in; (void)out_size; (void)ws_size;

  const float* x         = (const float*)d_in[0];
  const float* in_proj_w = (const float*)d_in[1];
  const float* conv_w    = (const float*)d_in[2];
  const float* conv_b    = (const float*)d_in[3];
  const float* norm_w    = (const float*)d_in[4];
  const float* out_proj_w= (const float*)d_in[5];
  const float* dt_bias   = (const float*)d_in[6];
  const float* A_log     = (const float*)d_in[7];
  const float* Dp        = (const float*)d_in[8];
  float* out = (float*)d_out;

  // workspace carve-out
  char* ws = (char*)d_ws;
  size_t off = 0;
  auto take = [&](size_t bytes) { char* p = ws + off; off = alignUp256(off + bytes); return p; };

  unsigned short* xw_bf16  = (unsigned short*)take((size_t)ROWS * D_MODEL * 2);
  unsigned short* wi_bf16  = (unsigned short*)take((size_t)D_MODEL * D_IN_PROJ * 2);
  unsigned short* wo_bf16  = (unsigned short*)take((size_t)D_INNER * D_MODEL * 2);
  float*          zxbcdt   = (float*)take((size_t)ROWS * D_IN_PROJ * 4);
  float*          xBC_act  = (float*)take((size_t)ROWS * CONV_DIM * 4);
  float*          dt_proc  = (float*)take((size_t)ROWS * NHEADS * 4);
  float*          y_buf    = (float*)take((size_t)ROWS * D_INNER * 4);
  unsigned short* g_bf16   = (unsigned short*)take((size_t)ROWS * D_INNER * 2);

  // 1) fp32 -> bf16 conversions
  {
    int n = ROWS * D_MODEL;
    cvt_bf16_kernel<<<(n + 255) / 256, 256, 0, stream>>>(x, xw_bf16, n);
  }
  {
    int n = D_MODEL * D_IN_PROJ;
    cvt_bf16_kernel<<<(n + 255) / 256, 256, 0, stream>>>(in_proj_w, wi_bf16, n);
  }
  {
    int n = D_INNER * D_MODEL;
    cvt_bf16_kernel<<<(n + 255) / 256, 256, 0, stream>>>(out_proj_w, wo_bf16, n);
  }

  // 2) in_proj GEMM: (8192x1024)x(1024x4384) -> zxbcdt
  {
    dim3 grid((D_IN_PROJ + 127) / 128, ROWS / 128);
    gemm_bf16_wmma<<<grid, 256, 0, stream>>>(xw_bf16, wi_bf16, zxbcdt,
                                             ROWS, D_IN_PROJ, D_MODEL);
  }

  // 3) depthwise causal conv + SiLU
  {
    int n = ROWS * CONV_DIM;
    conv_silu_kernel<<<(n + 255) / 256, 256, 0, stream>>>(zxbcdt, conv_w, conv_b, xBC_act);
  }

  // 4) dt softplus
  {
    int n = ROWS * NHEADS;
    dt_softplus_kernel<<<(n + 255) / 256, 256, 0, stream>>>(zxbcdt, dt_bias, dt_proc);
  }

  // 5) selective scan (one WG per (b,h))
  scan_kernel<<<BATCH * NHEADS, 256, 0, stream>>>(xBC_act, dt_proc, A_log, Dp, y_buf);

  // 6) gated RMSNorm -> bf16
  gate_rmsnorm_kernel<<<ROWS, 256, 0, stream>>>(y_buf, zxbcdt, norm_w, g_bf16);

  // 7) out_proj GEMM: (8192x2048)x(2048x1024) -> out (f32)
  {
    dim3 grid(D_MODEL / 128, ROWS / 128);
    gemm_bf16_wmma<<<grid, 256, 0, stream>>>(g_bf16, wo_bf16, out,
                                             ROWS, D_MODEL, D_INNER);
  }
}
